// SimpleRNN_22660247454201
// MI455X (gfx1250) — matrix operand order repaired and checked by reading, not compiled, not run on hardware
//
#include <hip/hip_runtime.h>
#include <hip/hip_bf16.h>
#include <math.h>

// ---------------------------------------------------------------------------
// Persistent GRU+ODE scan kernel for MI455X (gfx1250, wave32, WMMA).
//
// Grid: 32 workgroups x 384 threads (12 waves). Each workgroup owns 16 batch
// rows (one WMMA M-tile) and iterates all K=1024 steps. Each wave owns 2 of
// the 24 output-column tiles of the 3H=384 GRU gate matrix; its bf16 WMMA
// B-fragments of W_ih / W_hh live in VGPRs for the whole kernel (~96 VGPRs),
// keeping per-wave allocation under 256 (no s_set_vgpr_msb in the hot loop).
// GEMMs use v_wmma_f32_16x16x32_bf16 with f32 accumulation; all recurrent
// state / gates / RK4 integration stays fp32.
// ---------------------------------------------------------------------------

typedef __bf16 bf16_t;
typedef bf16_t bf16x16 __attribute__((ext_vector_type(16)));
typedef float  f32x8   __attribute__((ext_vector_type(8)));

#define B_TOT    512
#define K_STEPS  1024
#define U_DIM    4
#define P_DIM    5
#define L_DIM    64
#define H_DIM    128
#define G3       384        // 3*H
#define ROWS     16         // batch rows per workgroup == WMMA M
#define NTHREADS 384        // 12 waves
#define TILES_PER_WAVE 2    // 24 N-tiles of 16 across 12 waves

__device__ __forceinline__ float sigf(float x) { return 1.0f / (1.0f + __expf(-x)); }

__global__ __launch_bounds__(NTHREADS, 1)
void rnn_ode_persistent(
    const float* __restrict__ y0,          // (512,5)
    const float* __restrict__ u_seq,       // (512,1024,4)
    const float* __restrict__ dt_seq,      // (512,1024)
    const float* __restrict__ W_lift,      // (64,9)
    const float* __restrict__ b_lift,      // (64,)
    const float* __restrict__ W_ih,        // (384,64)
    const float* __restrict__ W_hh,        // (384,128)
    const float* __restrict__ b_ih,        // (384,)
    const float* __restrict__ b_hh,        // (384,)
    const float* __restrict__ W_head,      // (13,128)
    const float* __restrict__ b_head,      // (13,)
    const float* __restrict__ u_to_y_jump, // (4,5)
    float* __restrict__ y_out,             // (512,1024,5)
    float* __restrict__ rates_out,         // (512,1024,8)
    float* __restrict__ dhid_out)          // (512,1024,5)
{
    // ---------------- LDS state (~57 KB) ----------------
    __shared__ float  sU[ROWS][U_DIM];        // u_k staging
    __shared__ float  sDT[ROWS];              // dt_k staging
    __shared__ float  sY[ROWS][P_DIM];        // ODE state y (fp32, persistent)
    __shared__ bf16_t sX[ROWS][L_DIM];        // lift output, bf16 for A-frags
    __shared__ bf16_t sHbf[ROWS][H_DIM];      // h state, bf16 copy for A-frags
    __shared__ float  sH[ROWS][H_DIM];        // h state fp32 (persistent)
    __shared__ float  sG[ROWS][G3];           // [0,256)=gi+gh(+biases), [256,384)=i_n(+b_ih)
    __shared__ float  sHN[ROWS][H_DIM];       // h_n part (+b_hh)
    __shared__ float  sWlift[L_DIM * 9];
    __shared__ float  sBlift[L_DIM];
    __shared__ float  sWhead[13 * H_DIM];
    __shared__ float  sBhead[13];
    __shared__ float  sUJ[20];                // u_to_y_jump, row-major (4,5)
    __shared__ float  sRaw[ROWS][13];         // head output

    const int tid    = threadIdx.x;
    const int lane   = tid & 31;
    const int wv     = tid >> 5;              // wave id 0..11
    const int colN   = lane & 15;             // WMMA N / B-column
    const int hiHalf = (lane >> 4) & 1;       // lane half selects K-subrange
    const int b0     = blockIdx.x * ROWS;

    // ---------------- one-time setup ----------------
    for (int i = tid; i < L_DIM * 9; i += NTHREADS) sWlift[i] = W_lift[i];
    for (int i = tid; i < L_DIM;     i += NTHREADS) sBlift[i] = b_lift[i];
    for (int i = tid; i < 13*H_DIM;  i += NTHREADS) sWhead[i] = W_head[i];
    if (tid < 13) sBhead[tid] = b_head[tid];
    if (tid < 20) sUJ[tid] = u_to_y_jump[tid];
    for (int i = tid; i < ROWS * P_DIM; i += NTHREADS) {
        int r = i / P_DIM, c = i % P_DIM;
        sY[r][c] = y0[(size_t)(b0 + r) * P_DIM + c] + 0.01f;   // y_init
    }
    for (int i = tid; i < ROWS * H_DIM; i += NTHREADS) {
        sH[i >> 7][i & 127]   = 0.0f;                           // h0 = 0
        sHbf[i >> 7][i & 127] = (bf16_t)0.0f;
    }

    // B-matrix WMMA fragments in registers (bf16), contiguous k order (same as A):
    //   b[j] = B[kc*32 + hiHalf*16 + j][colN],  B[k][n] = W[n][k]
    // Per-lane loop-invariant biases hoisted into registers too.
    bf16x16 Bih[TILES_PER_WAVE][2];   // K = 64  -> 2 k-chunks of 32
    bf16x16 Bhh[TILES_PER_WAVE][4];   // K = 128 -> 4 k-chunks of 32
    float   biasI[TILES_PER_WAVE];    // b_ih[n] for this lane's column
    float   biasH[TILES_PER_WAVE];    // b_hh[n] for this lane's column
#pragma unroll
    for (int t = 0; t < TILES_PER_WAVE; ++t) {
        const int n = (wv * TILES_PER_WAVE + t) * 16 + colN;   // output column
        biasI[t] = b_ih[n];
        biasH[t] = b_hh[n];
#pragma unroll
        for (int kc = 0; kc < 2; ++kc)
#pragma unroll
            for (int j = 0; j < 16; ++j) {
                int kk = kc * 32 + hiHalf * 16 + j;
                Bih[t][kc][j] = (bf16_t)W_ih[(size_t)n * L_DIM + kk];
            }
#pragma unroll
        for (int kc = 0; kc < 4; ++kc)
#pragma unroll
            for (int j = 0; j < 16; ++j) {
                int kk = kc * 32 + hiHalf * 16 + j;
                Bhh[t][kc][j] = (bf16_t)W_hh[(size_t)n * H_DIM + kk];
            }
    }
    __syncthreads();

    // A-fragment lane mapping (contiguous k order, same as B):
    //   row M = lane&15; element j -> K = kbase + j, kbase = hiHalf*16
    const int rowM  = lane & 15;
    const int kbase = hiHalf * 16;

    // ---------------- time-step loop ----------------
    for (int k = 0; k < K_STEPS; ++k) {
        // --- stage 1: stage u_k, dt_k into LDS; prefetch next step's u ---
        if (tid < ROWS * U_DIM) {
            int r = tid >> 2, j = tid & 3;
            sU[r][j] = u_seq[((size_t)(b0 + r) * K_STEPS + k) * U_DIM + j];
            if (k + 1 < K_STEPS)
                __builtin_prefetch(&u_seq[((size_t)(b0 + r) * K_STEPS + (k + 1)) * U_DIM + j], 0, 1);
        }
        if (tid < ROWS) sDT[tid] = dt_seq[(size_t)(b0 + tid) * K_STEPS + k];
        __syncthreads();

        // --- stage 2: lift  x = silu([u,y] @ W_lift.T + b) -> bf16 ---
        for (int e = tid; e < ROWS * L_DIM; e += NTHREADS) {
            int r = e >> 6, c = e & 63;
            float acc = sBlift[c];
#pragma unroll
            for (int j = 0; j < U_DIM; ++j) acc += sU[r][j] * sWlift[c * 9 + j];
#pragma unroll
            for (int j = 0; j < P_DIM; ++j) acc += sY[r][j] * sWlift[c * 9 + U_DIM + j];
            float xv = acc * sigf(acc);                         // silu
            sX[r][c] = (bf16_t)xv;
        }
        __syncthreads();

        // --- stage 3: WMMA  gi = x@W_ih.T, gh = h@W_hh.T ---
        bf16x16 ax[2], ah[4];
#pragma unroll
        for (int kc = 0; kc < 2; ++kc)
#pragma unroll
            for (int j = 0; j < 16; ++j)
                ax[kc][j] = sX[rowM][kc * 32 + kbase + j];
#pragma unroll
        for (int kc = 0; kc < 4; ++kc)
#pragma unroll
            for (int j = 0; j < 16; ++j)
                ah[kc][j] = sHbf[rowM][kc * 32 + kbase + j];

#pragma unroll
        for (int t = 0; t < TILES_PER_WAVE; ++t) {
            const int n0 = (wv * TILES_PER_WAVE + t) * 16;
            const int n  = n0 + colN;
            f32x8 acc_i = {};
            f32x8 acc_h = {};
#pragma unroll
            for (int kc = 0; kc < 2; ++kc)
                acc_i = __builtin_amdgcn_wmma_f32_16x16x32_bf16(
                    false, ax[kc], false, Bih[t][kc], (short)0, acc_i, false, false);
#pragma unroll
            for (int kc = 0; kc < 4; ++kc)
                acc_h = __builtin_amdgcn_wmma_f32_16x16x32_bf16(
                    false, ah[kc], false, Bhh[t][kc], (short)0, acc_h, false, false);

            // C/D layout: c[v] is (M = v + 8*hiHalf, N = colN)
            if (n0 < 2 * H_DIM) {             // r,z gates: gi+gh fused
                const float bsum = biasI[t] + biasH[t];
#pragma unroll
                for (int v = 0; v < 8; ++v) {
                    int M = v + hiHalf * 8;
                    sG[M][n] = acc_i[v] + acc_h[v] + bsum;
                }
            } else {                          // n gate: keep i_n / h_n separate
#pragma unroll
                for (int v = 0; v < 8; ++v) {
                    int M = v + hiHalf * 8;
                    sG[M][n]            = acc_i[v] + biasI[t];
                    sHN[M][n - 2*H_DIM] = acc_h[v] + biasH[t];
                }
            }
        }
        __syncthreads();

        // --- stage 4: GRU elementwise update (fp32) ---
        for (int e = tid; e < ROWS * H_DIM; e += NTHREADS) {
            int r = e >> 7, c = e & 127;
            float rg = sigf(sG[r][c]);
            float zg = sigf(sG[r][H_DIM + c]);
            float nn = tanhf(sG[r][2 * H_DIM + c] + rg * sHN[r][c]);
            float hn = (1.0f - zg) * nn + zg * sH[r][c];
            sH[r][c]   = hn;
            sHbf[r][c] = (bf16_t)hn;          // bf16 copy for next step's WMMA
        }
        __syncthreads();

        // --- stage 5: head  raw = h_new @ W_head.T + b_head ---
        if (tid < ROWS * 13) {
            int r = tid / 13, o = tid % 13;
            float acc = sBhead[o];
#pragma unroll 8
            for (int c = 0; c < H_DIM; ++c) acc += sH[r][c] * sWhead[o * H_DIM + c];
            sRaw[r][o] = acc;
        }
        __syncthreads();

        // --- stage 6: rates/dhid, y_jump, 10x RK4, outputs (1 lane / row) ---
        if (tid < ROWS) {
            const int r = tid;
            float rates[8], dh[P_DIM], y[P_DIM];
#pragma unroll
            for (int i = 0; i < 8; ++i) rates[i] = 0.01f + 2.99f * sigf(sRaw[r][i]);
#pragma unroll
            for (int i = 0; i < P_DIM; ++i) dh[i] = 3.0f * sigf(sRaw[r][8 + i]);
#pragma unroll
            for (int i = 0; i < P_DIM; ++i) {
                float v = sY[r][i] + dh[i];
#pragma unroll
                for (int j = 0; j < U_DIM; ++j) v += sU[r][j] * sUJ[j * P_DIM + i];
                y[i] = v;
            }
            const float kf1 = rates[0], kf2 = rates[1], kf3 = rates[2], kf4 = rates[3];
            const float kr1 = rates[4], kr2 = rates[5], kr3 = rates[6], kr4 = rates[7];
            const float hs = sDT[r] * 0.1f;

#define CHAIN_RHS(Y, D)                                                     \
            D[0] = -kf1 * Y[0] + kr1 * Y[1];                                \
            D[1] =  kf1 * Y[0] - kr1 * Y[1] - kf2 * Y[1] + kr2 * Y[2];      \
            D[2] =  kf2 * Y[1] - kr2 * Y[2] - kf3 * Y[2] + kr3 * Y[3];      \
            D[3] =  kf3 * Y[2] - kr3 * Y[3] - kf4 * Y[3] + kr4 * Y[4];      \
            D[4] =  kf4 * Y[3] - kr4 * Y[4];

            for (int s = 0; s < 10; ++s) {
                float k1[P_DIM], k2[P_DIM], k3[P_DIM], k4[P_DIM], tmp[P_DIM];
                CHAIN_RHS(y, k1)
#pragma unroll
                for (int i = 0; i < P_DIM; ++i) tmp[i] = y[i] + 0.5f * hs * k1[i];
                CHAIN_RHS(tmp, k2)
#pragma unroll
                for (int i = 0; i < P_DIM; ++i) tmp[i] = y[i] + 0.5f * hs * k2[i];
                CHAIN_RHS(tmp, k3)
#pragma unroll
                for (int i = 0; i < P_DIM; ++i) tmp[i] = y[i] + hs * k3[i];
                CHAIN_RHS(tmp, k4)
#pragma unroll
                for (int i = 0; i < P_DIM; ++i)
                    y[i] = fmaxf(y[i] + (hs * (1.0f / 6.0f)) *
                                        (k1[i] + 2.0f * k2[i] + 2.0f * k3[i] + k4[i]), 0.0f);
            }
#undef CHAIN_RHS

            const size_t base = (size_t)(b0 + r) * K_STEPS + k;
#pragma unroll
            for (int i = 0; i < P_DIM; ++i) { sY[r][i] = y[i]; y_out[base * P_DIM + i] = y[i]; }
#pragma unroll
            for (int i = 0; i < 8; ++i) rates_out[base * 8 + i] = rates[i];
#pragma unroll
            for (int i = 0; i < P_DIM; ++i) dhid_out[base * P_DIM + i] = dh[i];
        }
        __syncthreads();   // protect sY/sU before next iteration rewrites them
    }
}

extern "C" void kernel_launch(void* const* d_in, const int* in_sizes, int n_in,
                              void* d_out, int out_size, void* d_ws, size_t ws_size,
                              hipStream_t stream) {
    const float* y0          = (const float*)d_in[0];
    const float* u_seq       = (const float*)d_in[1];
    const float* dt_seq      = (const float*)d_in[2];
    const float* W_lift      = (const float*)d_in[3];
    const float* b_lift      = (const float*)d_in[4];
    const float* W_ih        = (const float*)d_in[5];
    const float* W_hh        = (const float*)d_in[6];
    const float* b_ih        = (const float*)d_in[7];
    const float* b_hh        = (const float*)d_in[8];
    const float* W_head      = (const float*)d_in[9];
    const float* b_head      = (const float*)d_in[10];
    const float* u_to_y_jump = (const float*)d_in[11];

    float* out       = (float*)d_out;
    float* y_out     = out;                                          // (B,K,5)
    float* rates_out = y_out + (size_t)B_TOT * K_STEPS * P_DIM;      // (B,K,8)
    float* dhid_out  = rates_out + (size_t)B_TOT * K_STEPS * 8;      // (B,K,5)

    dim3 grid(B_TOT / ROWS);   // 32 workgroups
    dim3 block(NTHREADS);      // 12 waves (wave32)
    rnn_ode_persistent<<<grid, block, 0, stream>>>(
        y0, u_seq, dt_seq, W_lift, b_lift, W_ih, W_hh, b_ih, b_hh,
        W_head, b_head, u_to_y_jump, y_out, rates_out, dhid_out);
}
